// MCNodeProcessor_58171037057130
// MI455X (gfx1250) — compile-verified, matrix operand order checked by reading, not run on hardware
//
#include <hip/hip_runtime.h>
#include <math.h>

// ---------- problem constants ----------
#define NNODES 262144
#define DDIM   128
#define EEDGES 524288
#define KPAD   288      // 2*D + 5 = 261 padded up to 9*32
#define KC1    9        // k-chunks of 32 for GEMM1
#define KC2    4        // k-chunks of 32 for GEMM2
#define NT     8        // 8 n-tiles of 16 -> D=128

typedef __attribute__((ext_vector_type(16))) __bf16          v16bf;
typedef __attribute__((ext_vector_type(8)))  float           v8f;
typedef __attribute__((ext_vector_type(4)))  float           v4f;
typedef __attribute__((ext_vector_type(4)))  unsigned int    v4u;
typedef __attribute__((ext_vector_type(4)))  unsigned short  v4s;

union FragBF { v4u u[2]; v16bf v; };

__device__ __forceinline__ unsigned short f2bf(float f) {
  unsigned int u = __float_as_uint(f);
  unsigned int r = (u + 0x7FFFu + ((u >> 16) & 1u)) >> 16;   // RNE
  return (unsigned short)r;
}
__device__ __forceinline__ float signed_log(float x) {
  float s = (x > 0.f) ? 1.f : ((x < 0.f) ? -1.f : 0.f);
  return s * logf(fabsf(x) + 1e-8f);
}

// ---------- kernel 1: pack W1/W2 into bf16 WMMA B-fragment layout ----------
// frag element address: ((kc*NT + nt)*32 + lane)*16 + i
// lane (0..15)  -> column n = nt*16 + lane,   K = {8g..8g+7, 16+8g..23+8g}, g = lane>>4
__global__ __launch_bounds__(256) void pack_weights(
    const float* __restrict__ W1, const float* __restrict__ W2,
    unsigned short* __restrict__ w1f, unsigned short* __restrict__ w2f) {
  const int W1TOT = KC1 * NT * 32 * 16;   // 36864
  const int W2TOT = KC2 * NT * 32 * 16;   // 16384
  int idx = blockIdx.x * 256 + threadIdx.x;
  if (idx < W1TOT) {
    int i    = idx & 15;
    int lane = (idx >> 4) & 31;
    int nt   = (idx >> 9) & 7;
    int kc   = idx >> 12;
    int g    = lane >> 4;
    int n    = nt * 16 + (lane & 15);
    int kl   = (i < 8) ? (8 * g + i) : (16 + 8 * g + (i - 8));
    int k    = kc * 32 + kl;
    float v  = (k < 2 * DDIM + 5) ? W1[k * DDIM + n] : 0.f;
    w1f[idx] = f2bf(v);
  } else if (idx < W1TOT + W2TOT) {
    int j    = idx - W1TOT;
    int i    = j & 15;
    int lane = (j >> 4) & 31;
    int nt   = (j >> 9) & 7;
    int kc   = j >> 12;
    int g    = lane >> 4;
    int n    = nt * 16 + (lane & 15);
    int kl   = (i < 8) ? (8 * g + i) : (16 + 8 * g + (i - 8));
    int k    = kc * 32 + kl;
    w2f[j]   = f2bf(W2[k * DDIM + n]);
  }
}

// ---------- kernel 2: zero upstream accumulator ----------
__global__ __launch_bounds__(256) void zero_up(float* __restrict__ up) {
  size_t i = ((size_t)blockIdx.x * 256 + threadIdx.x) * 4;
  if (i < (size_t)NNODES * DDIM) {
    v4f z = {};
    *(v4f*)(up + i) = z;
  }
}

// ---------- kernel 3: edge scatter-add (one wave per edge) ----------
__global__ __launch_bounds__(256) void scatter_edges(
    const float* __restrict__ h, const int* __restrict__ src,
    const int* __restrict__ dst, float* __restrict__ up) {
  unsigned int idx = blockIdx.x * 256u + threadIdx.x;    // E*32 threads
  unsigned int e = idx >> 5;
  if (e >= EEDGES) return;
  int col = (idx & 31) << 2;
  int s = src[e], d = dst[e];
  v4f hv = *(const v4f*)(h + (size_t)s * DDIM + col);
  float* p = up + (size_t)d * DDIM + col;
  atomicAdd(p + 0, hv.x);
  atomicAdd(p + 1, hv.y);
  atomicAdd(p + 2, hv.z);
  atomicAdd(p + 3, hv.w);
}

// ---------- kernel 4: fused MLP (bf16 WMMA) + residual + LayerNorm ----------
__global__ __launch_bounds__(128) void fused_mlp_ln(
    const float* __restrict__ h,  const float* __restrict__ up,
    const float* __restrict__ c1, const float* __restrict__ c2,
    const float* __restrict__ c3, const float* __restrict__ c4,
    const float* __restrict__ qn,
    const unsigned short* __restrict__ w1f, const unsigned short* __restrict__ w2f,
    const float* __restrict__ b1, const float* __restrict__ b2,
    const float* __restrict__ gamma, const float* __restrict__ beta,
    float* __restrict__ out) {
  __shared__ __align__(16) unsigned short Abuf[64 * KPAD];  // 36864 B
  __shared__ __align__(16) unsigned short Hbuf[64 * DDIM];  // 16384 B
  __shared__ float b1s[DDIM], b2s[DDIM], gs[DDIM], bs[DDIM];

  const int tid  = threadIdx.x;
  const int lane = tid & 31;
  const int wave = tid >> 5;
  const int g    = lane >> 4;          // lane group (K-half select)
  const int base = blockIdx.x * 64;    // first node of this block

  if (tid < DDIM) { b1s[tid] = b1[tid]; b2s[tid] = b2[tid]; gs[tid] = gamma[tid]; bs[tid] = beta[tid]; }

  // ---- build bf16 A-tile: [h | upstream_h | signed_log(phys) | zero-pad] ----
  for (int idx = tid; idx < 64 * 32; idx += 128) {
    int nl  = idx >> 5;
    int col = (idx & 31) << 2;
    v4f hv = *(const v4f*)(h  + (size_t)(base + nl) * DDIM + col);
    v4f uv = *(const v4f*)(up + (size_t)(base + nl) * DDIM + col);
    v4s ph = { f2bf(hv.x), f2bf(hv.y), f2bf(hv.z), f2bf(hv.w) };
    v4s pu = { f2bf(uv.x), f2bf(uv.y), f2bf(uv.z), f2bf(uv.w) };
    *(v4s*)(&Abuf[nl * KPAD + col])        = ph;
    *(v4s*)(&Abuf[nl * KPAD + DDIM + col]) = pu;
  }
  if (tid < 64) {
    int node = base + tid;
    unsigned short* p = &Abuf[tid * KPAD + 2 * DDIM];
    p[0] = f2bf(signed_log(c1[node]));
    p[1] = f2bf(signed_log(c2[node]));
    p[2] = f2bf(signed_log(c3[node]));
    p[3] = f2bf(signed_log(c4[node]));
    p[4] = f2bf(signed_log(qn[node]));
#pragma unroll
    for (int i = 5; i < 32; ++i) p[i] = 0;
  }
  __syncthreads();

  // ---- GEMM1: (16 x 288) @ (288 x 128), bf16 WMMA, f32 acc ----
  const int mrow = wave * 16 + (lane & 15);   // A row held by this lane
  v8f acc[NT];
#pragma unroll
  for (int nt = 0; nt < NT; ++nt) acc[nt] = (v8f){};
  for (int kc = 0; kc < KC1; ++kc) {
    FragBF fa;
    const unsigned short* pa = &Abuf[mrow * KPAD + kc * 32 + 8 * g];
    fa.u[0] = *(const v4u*)(pa);        // K = kc*32 + 8g .. +7
    fa.u[1] = *(const v4u*)(pa + 16);   // K = kc*32 + 16 + 8g .. +7
#pragma unroll
    for (int nt = 0; nt < NT; ++nt) {
      FragBF fb;
      const unsigned short* pb = w1f + (((size_t)(kc * NT + nt) * 32 + lane) * 16);
      fb.u[0] = *(const v4u*)(pb);
      fb.u[1] = *(const v4u*)(pb + 8);
      acc[nt] = __builtin_amdgcn_wmma_f32_16x16x32_bf16(
          false, fa.v, false, fb.v, (short)0, acc[nt], false, false);
    }
  }

  // ---- bias + SiLU -> bf16 hid tile in LDS ----
  const int mbase = wave * 16;
  const int ncol  = lane & 15;
#pragma unroll
  for (int nt = 0; nt < NT; ++nt) {
    int n = nt * 16 + ncol;
    float bb = b1s[n];
#pragma unroll
    for (int v = 0; v < 8; ++v) {
      int m = v + 8 * g;                 // C layout: lanes 16-31 hold M = v+8
      float x = acc[nt][v] + bb;
      float sg = 1.f / (1.f + __expf(-x));
      Hbuf[(mbase + m) * DDIM + n] = f2bf(x * sg);
    }
  }
  __syncthreads();

  // ---- GEMM2: (16 x 128) @ (128 x 128) ----
  v8f acc2[NT];
#pragma unroll
  for (int nt = 0; nt < NT; ++nt) acc2[nt] = (v8f){};
  for (int kc = 0; kc < KC2; ++kc) {
    FragBF fa;
    const unsigned short* pa = &Hbuf[mrow * DDIM + kc * 32 + 8 * g];
    fa.u[0] = *(const v4u*)(pa);
    fa.u[1] = *(const v4u*)(pa + 16);
#pragma unroll
    for (int nt = 0; nt < NT; ++nt) {
      FragBF fb;
      const unsigned short* pb = w2f + (((size_t)(kc * NT + nt) * 32 + lane) * 16);
      fb.u[0] = *(const v4u*)(pb);
      fb.u[1] = *(const v4u*)(pb + 8);
      acc2[nt] = __builtin_amdgcn_wmma_f32_16x16x32_bf16(
          false, fa.v, false, fb.v, (short)0, acc2[nt], false, false);
    }
  }

  // ---- residual z = h + mlp_out + b2, per-row stats ----
  float s1[8], s2[8];
#pragma unroll
  for (int v = 0; v < 8; ++v) { s1[v] = 0.f; s2[v] = 0.f; }
#pragma unroll
  for (int nt = 0; nt < NT; ++nt) {
    int n = nt * 16 + ncol;
#pragma unroll
    for (int v = 0; v < 8; ++v) {
      int m = v + 8 * g;
      int node = base + mbase + m;
      float z = acc2[nt][v] + b2s[n] + h[(size_t)node * DDIM + n];
      acc2[nt][v] = z;
      s1[v] += z;
      s2[v] += z * z;
    }
  }
  // reduce across the 16 lanes holding one row (xor masks 1,2,4,8 stay in-group)
#pragma unroll
  for (int off = 1; off < 16; off <<= 1) {
#pragma unroll
    for (int v = 0; v < 8; ++v) {
      s1[v] += __shfl_xor(s1[v], off, 32);
      s2[v] += __shfl_xor(s2[v], off, 32);
    }
  }
  float mean[8], inv[8];
#pragma unroll
  for (int v = 0; v < 8; ++v) {
    mean[v] = s1[v] * (1.f / DDIM);
    float var = s2[v] * (1.f / DDIM) - mean[v] * mean[v];
    inv[v] = rsqrtf(var + 1e-5f);
  }

  // ---- normalize, scale/shift, store ----
#pragma unroll
  for (int nt = 0; nt < NT; ++nt) {
    int n = nt * 16 + ncol;
    float gm = gs[n], bt = bs[n];
#pragma unroll
    for (int v = 0; v < 8; ++v) {
      int m = v + 8 * g;
      int node = base + mbase + m;
      out[(size_t)node * DDIM + n] = (acc2[nt][v] - mean[v]) * inv[v] * gm + bt;
    }
  }
}

// ---------- host-side launch ----------
extern "C" void kernel_launch(void* const* d_in, const int* in_sizes, int n_in,
                              void* d_out, int out_size, void* d_ws, size_t ws_size,
                              hipStream_t stream) {
  const float* h     = (const float*)d_in[0];
  const float* c1    = (const float*)d_in[1];
  const float* c2    = (const float*)d_in[2];
  const float* c3    = (const float*)d_in[3];
  const float* c4    = (const float*)d_in[4];
  const float* qn    = (const float*)d_in[5];
  const int*   src   = (const int*)d_in[6];
  const int*   dst   = (const int*)d_in[7];
  const float* W1    = (const float*)d_in[8];
  const float* b1    = (const float*)d_in[9];
  const float* W2    = (const float*)d_in[10];
  const float* b2    = (const float*)d_in[11];
  const float* gamma = (const float*)d_in[12];
  const float* beta  = (const float*)d_in[13];
  float*       out   = (float*)d_out;

  // workspace layout
  float* up = (float*)d_ws;                                      // N*D f32 = 128 MiB
  unsigned short* w1f = (unsigned short*)((char*)d_ws + (size_t)NNODES * DDIM * 4);
  unsigned short* w2f = w1f + KC1 * NT * 32 * 16;

  const int W1TOT = KC1 * NT * 32 * 16, W2TOT = KC2 * NT * 32 * 16;
  pack_weights<<<(W1TOT + W2TOT + 255) / 256, 256, 0, stream>>>(W1, W2, w1f, w2f);
  zero_up<<<(int)(((size_t)NNODES * DDIM / 4 + 255) / 256), 256, 0, stream>>>(up);
  scatter_edges<<<(EEDGES * 32) / 256, 256, 0, stream>>>(h, src, dst, up);
  fused_mlp_ln<<<NNODES / 64, 128, 0, stream>>>(h, up, c1, c2, c3, c4, qn,
                                                w1f, w2f, b1, b2, gamma, beta, out);
}
